// RecurrentModel_76424648065501
// MI455X (gfx1250) — compile-verified, matrix operand order checked by reading, not run on hardware
//
#include <hip/hip_runtime.h>
#include <hip/hip_bf16.h>

#define T_   256
#define B_   256
#define IN_  64
#define HID_ 512
#define LS_  256
#define G3_  768  // 3*LS

typedef __bf16 bf16;
typedef __attribute__((ext_vector_type(16))) __bf16 v16bf;
typedef __attribute__((ext_vector_type(8)))  __bf16 v8bf;
typedef __attribute__((ext_vector_type(8)))  float  v8f;

constexpr int NWG      = 96;
constexpr int TPB      = 256;                 // 8 wave32 per workgroup
constexpr int NWAVES   = NWG * (TPB / 32);    // 768 waves
constexpr int NTHREADS = NWG * TPB;

// ---------------------------------------------------------------- prep kernels

__global__ void k_f32_to_bf16(const float* __restrict__ src, bf16* __restrict__ dst, int n) {
  int i = blockIdx.x * blockDim.x + threadIdx.x;
  if (i < n) dst[i] = (bf16)src[i];
}

__global__ void k_prep_state(const float* __restrict__ h,
                             float* __restrict__ sf, bf16* __restrict__ sb,
                             unsigned* __restrict__ bar) {
  int i = blockIdx.x * blockDim.x + threadIdx.x;
  if (i < B_ * HID_) { float v = h[i]; sf[i] = v; sb[i] = (bf16)v; }
  if (i < 8) bar[i] = 0u;
}

// Wc = w_ih0 @ W_lin  ([768,512]@[512,64] -> [768,64]), bc = b_ih0 + w_ih0 @ b_lin
__global__ void k_prep_wc(const float* __restrict__ w_ih0, const float* __restrict__ W_lin,
                          const float* __restrict__ b_ih0, const float* __restrict__ b_lin,
                          bf16* __restrict__ wc, float* __restrict__ bc) {
  int i = blockIdx.x * blockDim.x + threadIdx.x;
  if (i >= G3_ * IN_) return;
  int j = i >> 6;      // output row 0..767
  int k = i & 63;      // output col 0..63
  const float* wr = w_ih0 + j * HID_;
  float acc = 0.f;
  for (int p = 0; p < HID_; ++p) acc += wr[p] * W_lin[p * IN_ + k];
  wc[j * IN_ + k] = (bf16)acc;
  if (k == 0) {
    float bacc = b_ih0[j];
    for (int p = 0; p < HID_; ++p) bacc += wr[p] * b_lin[p];
    bc[j] = bacc;
  }
}

// ---------------------------------------------------------------- device utils

__device__ __forceinline__ float sigmoidf_(float x) { return 1.f / (1.f + __expf(-x)); }

__device__ __forceinline__ void grid_sync(unsigned* cnt, unsigned* gen) {
  __threadfence();
  __syncthreads();
  if (threadIdx.x == 0) {
    unsigned g = __hip_atomic_load(gen, __ATOMIC_ACQUIRE, __HIP_MEMORY_SCOPE_AGENT);
    unsigned arrived =
        __hip_atomic_fetch_add(cnt, 1u, __ATOMIC_ACQ_REL, __HIP_MEMORY_SCOPE_AGENT);
    if (arrived == (unsigned)(NWG - 1)) {
      __hip_atomic_store(cnt, 0u, __ATOMIC_RELAXED, __HIP_MEMORY_SCOPE_AGENT);
      __hip_atomic_fetch_add(gen, 1u, __ATOMIC_ACQ_REL, __HIP_MEMORY_SCOPE_AGENT);
    } else {
      while (__hip_atomic_load(gen, __ATOMIC_ACQUIRE, __HIP_MEMORY_SCOPE_AGENT) == g) {
        __builtin_amdgcn_s_sleep(1);
      }
    }
  }
  __syncthreads();
  __threadfence();
}

// One wave computes a 16 x (16*NT) strip of  C[M,N] = mask_row * (A[M,K] * Bw[N,K]^T) + bias.
// A-frag: lane row = mT+(lane&15); K chunk {0..7,16..23} (lanes<16) / {8..15,24..31}.
// B-frag: lane col = nT+(lane&15)+16u; same K packing (column-per-lane).
// D: acc[u][r] -> row mT + r + 8*(lane>=16), col nT + 16u + (lane&15).
// Row mask (reset) is applied in the epilogue: (m*h) @ W^T == m * (h @ W^T).
template <int K, int NT>
__device__ __forceinline__ void mm_tile(const bf16* __restrict__ A, int lda,
                                        const bf16* __restrict__ Bw, int ldb,
                                        float* __restrict__ C, int ldc,
                                        const float* __restrict__ bias,
                                        int mT, int nT,
                                        const unsigned char* __restrict__ rst) {
  const int lane = threadIdx.x & 31;
  const int half = lane >> 4;
  const int l15  = lane & 15;
  const bf16* ap  = A + (mT + l15) * lda + half * 8;
  const bf16* bp0 = Bw + (nT + l15) * ldb + half * 8;
  v8f acc[NT];
#pragma unroll
  for (int u = 0; u < NT; ++u) acc[u] = (v8f){};
#pragma unroll
  for (int kk = 0; kk < K; kk += 32) {
    v8bf alo = *(const v8bf*)(ap + kk);
    v8bf ahi = *(const v8bf*)(ap + kk + 16);
    v16bf af = __builtin_shufflevector(alo, ahi, 0, 1, 2, 3, 4, 5, 6, 7,
                                       8, 9, 10, 11, 12, 13, 14, 15);
#pragma unroll
    for (int u = 0; u < NT; ++u) {
      const bf16* bp = bp0 + u * 16 * ldb;
      v8bf blo = *(const v8bf*)(bp + kk);
      v8bf bhi = *(const v8bf*)(bp + kk + 16);
      v16bf bfv = __builtin_shufflevector(blo, bhi, 0, 1, 2, 3, 4, 5, 6, 7,
                                          8, 9, 10, 11, 12, 13, 14, 15);
      acc[u] = __builtin_amdgcn_wmma_f32_16x16x32_bf16(false, af, false, bfv,
                                                       (short)0, acc[u], false, false);
    }
  }
  float mrow[8];
#pragma unroll
  for (int r = 0; r < 8; ++r) {
    mrow[r] = (rst != nullptr && rst[mT + half * 8 + r] != 0) ? 0.f : 1.f;
  }
#pragma unroll
  for (int u = 0; u < NT; ++u) {
    const int ncol = nT + u * 16 + l15;
    const float bv = bias[ncol];
#pragma unroll
    for (int r = 0; r < 8; ++r) {
      C[(mT + half * 8 + r) * ldc + ncol] = acc[u][r] * mrow[r] + bv;
    }
  }
}

// ---------------------------------------------------------------- main scan

__global__ __launch_bounds__(TPB) void k_gru_scan(
    const bf16* __restrict__ obs_bf, const unsigned char* __restrict__ reset,
    const bf16* __restrict__ whh0, const bf16* __restrict__ wih1,
    const bf16* __restrict__ whh1, const bf16* __restrict__ wc,
    const float* __restrict__ bc, const float* __restrict__ b_hh0,
    const float* __restrict__ b_ih1, const float* __restrict__ b_hh1,
    float* __restrict__ state_f, bf16* __restrict__ state_b,
    float* __restrict__ hg0, float* __restrict__ hg1,
    float* __restrict__ ig0s, float* __restrict__ ig1s,
    float* __restrict__ out, unsigned* __restrict__ bar) {
  unsigned* cnt = bar;
  unsigned* gen = bar + 1;
  const int gtid  = blockIdx.x * TPB + threadIdx.x;
  const int gwave = gtid >> 5;

  for (int t = 0; t < T_; ++t) {
    const unsigned char* rst = reset + t * B_;

    // ---- Phase A: hg0 = m*(h0@whh0^T)+b, hg1 = m*(h1@whh1^T)+b, ig0 = obs_t@Wc^T+bc
    // 3 GEMMs x 256 strip-jobs (16x48 strips) = 768 jobs = exactly 1 per wave.
    for (int job = gwave; job < 3 * 256; job += NWAVES) {
      int which = job >> 8;
      int tid   = job & 255;
      int mT = (tid >> 4) * 16, nT = (tid & 15) * 48;
      if (which == 0)
        mm_tile<256, 3>(state_b, HID_, whh0, LS_, hg0, G3_, b_hh0, mT, nT, rst);
      else if (which == 1)
        mm_tile<256, 3>(state_b + LS_, HID_, whh1, LS_, hg1, G3_, b_hh1, mT, nT, rst);
      else
        mm_tile<64, 3>(obs_bf + (size_t)t * B_ * IN_, IN_, wc, IN_, ig0s, G3_, bc,
                       mT, nT, nullptr);
    }
    grid_sync(cnt, gen);

    // ---- Phase B: GRU cell 0 elementwise -> h0n
    for (int i = gtid; i < B_ * LS_; i += NTHREADS) {
      int b = i >> 8, j = i & 255;
      float m = rst[b] ? 0.f : 1.f;
      float r = sigmoidf_(ig0s[b * G3_ + j] + hg0[b * G3_ + j]);
      float z = sigmoidf_(ig0s[b * G3_ + LS_ + j] + hg0[b * G3_ + LS_ + j]);
      float n = tanhf(ig0s[b * G3_ + 2 * LS_ + j] + r * hg0[b * G3_ + 2 * LS_ + j]);
      float hp = state_f[b * HID_ + j] * m;
      float hn = (1.f - z) * n + z * hp;
      out[(size_t)t * B_ * HID_ + b * HID_ + j] = hn;
      state_f[b * HID_ + j] = hn;
      state_b[b * HID_ + j] = (bf16)hn;
    }
    grid_sync(cnt, gen);

    // ---- Phase C: ig1 = h0n @ wih1^T + b_ih1 (768 16x16 tiles = 1 per wave)
    for (int job = gwave; job < 768; job += NWAVES) {
      int mT = (job / 48) * 16, nT = (job % 48) * 16;
      mm_tile<256, 1>(state_b, HID_, wih1, LS_, ig1s, G3_, b_ih1, mT, nT, nullptr);
    }
    grid_sync(cnt, gen);

    // ---- Phase D: GRU cell 1 elementwise -> h1n
    for (int i = gtid; i < B_ * LS_; i += NTHREADS) {
      int b = i >> 8, j = i & 255;
      float m = rst[b] ? 0.f : 1.f;
      float r = sigmoidf_(ig1s[b * G3_ + j] + hg1[b * G3_ + j]);
      float z = sigmoidf_(ig1s[b * G3_ + LS_ + j] + hg1[b * G3_ + LS_ + j]);
      float n = tanhf(ig1s[b * G3_ + 2 * LS_ + j] + r * hg1[b * G3_ + 2 * LS_ + j]);
      float hp = state_f[b * HID_ + LS_ + j] * m;
      float hn = (1.f - z) * n + z * hp;
      out[(size_t)t * B_ * HID_ + b * HID_ + LS_ + j] = hn;
      state_f[b * HID_ + LS_ + j] = hn;
      state_b[b * HID_ + LS_ + j] = (bf16)hn;
    }
    grid_sync(cnt, gen);
  }
}

// ---------------------------------------------------------------- launch

extern "C" void kernel_launch(void* const* d_in, const int* in_sizes, int n_in,
                              void* d_out, int out_size, void* d_ws, size_t ws_size,
                              hipStream_t stream) {
  const float*         obs    = (const float*)d_in[0];
  const unsigned char* reset  = (const unsigned char*)d_in[1];  // numpy bool = 1 byte
  const float*         h0     = (const float*)d_in[2];
  const float*         W_lin  = (const float*)d_in[3];
  const float*         b_lin  = (const float*)d_in[4];
  const float*         w_ih0  = (const float*)d_in[5];
  const float*         w_hh0  = (const float*)d_in[6];
  const float*         b_ih0  = (const float*)d_in[7];
  const float*         b_hh0  = (const float*)d_in[8];
  const float*         w_ih1  = (const float*)d_in[9];
  const float*         w_hh1  = (const float*)d_in[10];
  const float*         b_ih1  = (const float*)d_in[11];
  const float*         b_hh1  = (const float*)d_in[12];
  float*               out    = (float*)d_out;

  char* ws = (char*)d_ws;
  size_t off = 0;
  auto carve = [&](size_t bytes) -> char* {
    char* p = ws + off;
    off = (off + bytes + 255) & ~(size_t)255;
    return p;
  };
  unsigned* bar      = (unsigned*)carve(64);
  bf16*  obs_bf      = (bf16*)carve((size_t)T_ * B_ * IN_ * 2);
  bf16*  whh0_bf     = (bf16*)carve((size_t)G3_ * LS_ * 2);
  bf16*  wih1_bf     = (bf16*)carve((size_t)G3_ * LS_ * 2);
  bf16*  whh1_bf     = (bf16*)carve((size_t)G3_ * LS_ * 2);
  bf16*  wc_bf       = (bf16*)carve((size_t)G3_ * IN_ * 2);
  float* bc          = (float*)carve((size_t)G3_ * 4);
  float* state_f     = (float*)carve((size_t)B_ * HID_ * 4);
  bf16*  state_b     = (bf16*)carve((size_t)B_ * HID_ * 2);
  float* hg0         = (float*)carve((size_t)B_ * G3_ * 4);
  float* hg1         = (float*)carve((size_t)B_ * G3_ * 4);
  float* ig0s        = (float*)carve((size_t)B_ * G3_ * 4);
  float* ig1s        = (float*)carve((size_t)B_ * G3_ * 4);
  (void)ws_size; (void)in_sizes; (void)n_in; (void)out_size;

  const int NW = G3_ * LS_;            // 196608 weight elems
  k_f32_to_bf16<<<(NW + 255) / 256, 256, 0, stream>>>(w_hh0, whh0_bf, NW);
  k_f32_to_bf16<<<(NW + 255) / 256, 256, 0, stream>>>(w_ih1, wih1_bf, NW);
  k_f32_to_bf16<<<(NW + 255) / 256, 256, 0, stream>>>(w_hh1, whh1_bf, NW);
  const int NO = T_ * B_ * IN_;        // 4194304 obs elems
  k_f32_to_bf16<<<(NO + 255) / 256, 256, 0, stream>>>(obs, obs_bf, NO);
  k_prep_state<<<(B_ * HID_ + 255) / 256, 256, 0, stream>>>(h0, state_f, state_b, bar);
  k_prep_wc<<<(G3_ * IN_ + 255) / 256, 256, 0, stream>>>(w_ih0, W_lin, b_ih0, b_lin,
                                                         wc_bf, bc);

  k_gru_scan<<<NWG, TPB, 0, stream>>>(obs_bf, reset, whh0_bf, wih1_bf, whh1_bf,
                                      wc_bf, bc, b_hh0, b_ih1, b_hh1,
                                      state_f, state_b, hg0, hg1, ig0s, ig1s,
                                      out, bar);
}
